// Model_6786048327840
// MI455X (gfx1250) — matrix operand order repaired and checked by reading, not compiled, not run on hardware
//
#include <hip/hip_runtime.h>
#include <hip/hip_bf16.h>

// ---------------------------------------------------------------------------
// CDNA5 (gfx1250) wave32 WMMA helpers: V_WMMA_F32_16X16X32_F16
// Fragment layouts per CDNA5 ISA 7.12.2 (16-bit A 16x32, 16-bit B 32x16,
// 32-bit C/D 16x16), wave32.
// ---------------------------------------------------------------------------
typedef _Float16 f16;
typedef __attribute__((ext_vector_type(16))) _Float16 v16h;
typedef __attribute__((ext_vector_type(8)))  float    v8f;

__device__ __forceinline__ v8f wmma16(v16h a, v16h b, v8f c) {
  return __builtin_amdgcn_wmma_f32_16x16x32_f16(false, a, false, b,
                                                (short)0, c, false, false);
}

// A fragment (16x32, MxK): lane L -> m = L%16; VGPR v holds K pair
// k = (v/4)*16 + (L/16)*8 + (v%4)*2
template <typename F>
__device__ __forceinline__ v16h load_a_frag(F f) {
  const int lane = threadIdx.x & 31;
  const int m = lane & 15;
  const int hh = lane >> 4;
  v16h a;
#pragma unroll
  for (int v = 0; v < 8; ++v) {
    int k = ((v >> 2) << 4) + (hh << 3) + ((v & 3) << 1);
    a[2 * v]     = f(m, k);
    a[2 * v + 1] = f(m, k + 1);
  }
  return a;
}

// B fragment (32x16, KxN): lane L -> n = L%16; VGPR v holds K pair
// k = (v/4)*16 + (L/16)*8 + (v%4)*2
template <typename F>
__device__ __forceinline__ v16h load_b_frag(F f) {
  const int lane = threadIdx.x & 31;
  const int n = lane & 15;
  const int hh = lane >> 4;
  v16h b;
#pragma unroll
  for (int v = 0; v < 8; ++v) {
    int k = ((v >> 2) << 4) + (hh << 3) + ((v & 3) << 1);
    b[2 * v]     = f(k, n);
    b[2 * v + 1] = f(k + 1, n);
  }
  return b;
}

// C/D (16x16 f32): lane L -> n = L%16; VGPR v holds row m = v + (L/16)*8
template <typename F>
__device__ __forceinline__ void store_c(v8f c, F f) {
  const int lane = threadIdx.x & 31;
  const int n = lane & 15;
  const int hh = lane >> 4;
#pragma unroll
  for (int v = 0; v < 8; ++v) f(v + (hh << 3), n, c[v]);
}

// half-wave (16-lane) reductions; rows of a C tile live within one 16-lane half
__device__ __forceinline__ float redmax16(float x) {
  x = fmaxf(x, __shfl_xor(x, 1, 32));
  x = fmaxf(x, __shfl_xor(x, 2, 32));
  x = fmaxf(x, __shfl_xor(x, 4, 32));
  x = fmaxf(x, __shfl_xor(x, 8, 32));
  return x;
}
__device__ __forceinline__ float redsum16(float x) {
  x += __shfl_xor(x, 1, 32);
  x += __shfl_xor(x, 2, 32);
  x += __shfl_xor(x, 4, 32);
  x += __shfl_xor(x, 8, 32);
  return x;
}

// ---------------------------------------------------------------------------
// CDNA5 async global->LDS bulk copy (ASYNCcnt-tracked, 16B per lane).
// VDST takes the LDS byte offset; ISA: flat LDS addr[31:0] == LDS offset,
// so truncating a generic pointer to a __shared__ object gives the operand.
// ---------------------------------------------------------------------------
__device__ __forceinline__ unsigned lds_off_u32(const void* p) {
  return (unsigned)(unsigned long long)p;
}
__device__ __forceinline__ void async_copy_g2l_b128(unsigned ldsoff,
                                                    const f16* gaddr) {
  asm volatile("global_load_async_to_lds_b128 %0, %1, off"
               :
               : "v"(ldsoff), "v"(gaddr)
               : "memory");
}
__device__ __forceinline__ void wait_async_zero() {
  asm volatile("s_wait_asynccnt 0x0" ::: "memory");
}

// ---------------------------------------------------------------------------
// K1: QKV projection with fused input transpose.
// x[s,i,e] = data[s,e,i]; qkv = x @ Wqkv^T + bqkv; scatter to Q/K/V f16 in
// [bm*4+head, s, d] layout. Grid (16384 row tiles, 12 col tiles), 1 wave.
// ---------------------------------------------------------------------------
__global__ void qkv_proj_kernel(const float* __restrict__ X,
                                const float* __restrict__ Wqkv,
                                const float* __restrict__ bqkv,
                                f16* __restrict__ Q, f16* __restrict__ K,
                                f16* __restrict__ V) {
  const int rt = blockIdx.x;
  const int ct = blockIdx.y;
  v8f acc = {};
#pragma unroll
  for (int kt = 0; kt < 2; ++kt) {
    v16h a = load_a_frag([&](int m, int k) -> f16 {
      int r = rt * 16 + m, s = r >> 8, bm = r & 255, e = kt * 32 + k;
      return (f16)X[((size_t)s * 64 + e) * 256 + bm];
    });
    v16h b = load_b_frag([&](int k, int n) -> f16 {
      return (f16)Wqkv[(size_t)(ct * 16 + n) * 64 + kt * 32 + k];
    });
    acc = wmma16(a, b, acc);
  }
  const int part = ct >> 2, head = ct & 3;
  f16* dst = (part == 0) ? Q : ((part == 1) ? K : V);
  store_c(acc, [&](int m, int n, float val) {
    int r = rt * 16 + m, s = r >> 8, bm = r & 255;
    val += bqkv[ct * 16 + n];
    dst[(((size_t)bm * 4 + head) * 1024 + s) * 16 + n] = (f16)val;
  });
}

// ---------------------------------------------------------------------------
// K2: single-wave flash attention per (bm*4+head, q-tile). hd=16 padded to
// K=32 in WMMA. Online softmax with half-wave shuffles; P routed via LDS to
// convert C-layout -> A-layout. Scores never hit memory.
// ---------------------------------------------------------------------------
__global__ void flash_attn_kernel(const f16* __restrict__ Q,
                                  const f16* __restrict__ K,
                                  const f16* __restrict__ V,
                                  f16* __restrict__ O) {
  const int bmh = blockIdx.x;  // 1024 = bm*4 + head
  const int qt = blockIdx.y;   // 64 q tiles
  const int bm = bmh >> 2, head = bmh & 3;
  __shared__ f16 Ps[16 * 16];
  const size_t base = (size_t)bmh * 1024 * 16;

  v16h qa = load_a_frag([&](int m, int k) -> f16 {
    return (k < 16)
               ? (f16)(0.25f * (float)Q[base + (size_t)(qt * 16 + m) * 16 + k])
               : (f16)0.0f;
  });
  v8f o = {};
  float mrow[8], lrow[8];
#pragma unroll
  for (int v = 0; v < 8; ++v) { mrow[v] = -3.0e38f; lrow[v] = 0.f; }

  const int lane = threadIdx.x & 31;
  const int cn = lane & 15, chh = lane >> 4;

  for (int kt = 0; kt < 64; ++kt) {
    v16h kb = load_b_frag([&](int k, int n) -> f16 {
      return (k < 16) ? K[base + (size_t)(kt * 16 + n) * 16 + k] : (f16)0.0f;
    });
    v8f s = {};
    s = wmma16(qa, kb, s);
#pragma unroll
    for (int v = 0; v < 8; ++v) {
      float tm = redmax16(s[v]);
      float mn = fmaxf(mrow[v], tm);
      float corr = __expf(mrow[v] - mn);
      float p = __expf(s[v] - mn);
      lrow[v] = lrow[v] * corr + redsum16(p);
      mrow[v] = mn;
      o[v] *= corr;
      Ps[(v + (chh << 3)) * 16 + cn] = (f16)p;
    }
    __syncthreads();
    v16h pa = load_a_frag([&](int m, int k) -> f16 {
      return (k < 16) ? Ps[m * 16 + k] : (f16)0.0f;
    });
    v16h vb = load_b_frag([&](int k, int n) -> f16 {
      return (k < 16) ? V[base + (size_t)(kt * 16 + k) * 16 + n] : (f16)0.0f;
    });
    o = wmma16(pa, vb, o);
    __syncthreads();
  }
  store_c(o, [&](int m, int n, float val) {
    int s = qt * 16 + m;
    float ov = val / lrow[m & 7];
    O[((size_t)s * 256 + bm) * 64 + head * 16 + n] = (f16)ov;
  });
}

// ---------------------------------------------------------------------------
// K3: attention output projection + fused transpose to emb[s,e,i].
// ---------------------------------------------------------------------------
__global__ void attn_outproj_kernel(const f16* __restrict__ O,
                                    const float* __restrict__ Wo,
                                    const float* __restrict__ bo,
                                    f16* __restrict__ emb) {
  const int rt = blockIdx.x;  // 16384 row tiles over S*Bm
  const int ct = blockIdx.y;  // 4 col tiles over E=64
  v8f acc = {};
#pragma unroll
  for (int kt = 0; kt < 2; ++kt) {
    v16h a = load_a_frag([&](int m, int k) -> f16 {
      return O[(size_t)(rt * 16 + m) * 64 + kt * 32 + k];
    });
    v16h b = load_b_frag([&](int k, int n) -> f16 {
      return (f16)Wo[(size_t)(ct * 16 + n) * 64 + kt * 32 + k];
    });
    acc = wmma16(a, b, acc);
  }
  store_c(acc, [&](int m, int n, float val) {
    int r = rt * 16 + m, s = r >> 8, i = r & 255, c = ct * 16 + n;
    emb[((size_t)s * 64 + c) * 256 + i] = (f16)(val + bo[c]);
  });
}

// ---------------------------------------------------------------------------
// K4: GRU input projection: [65536,256] @ Wih^T [256,192] + bih -> f16
// ---------------------------------------------------------------------------
__global__ void gru_xproj_kernel(const f16* __restrict__ emb,
                                 const float* __restrict__ Wih,
                                 const float* __restrict__ bih,
                                 f16* __restrict__ xp) {
  const int rt = blockIdx.x;  // 4096
  const int ct = blockIdx.y;  // 12
  v8f acc = {};
  for (int kt = 0; kt < 8; ++kt) {
    v16h a = load_a_frag([&](int m, int k) -> f16 {
      return emb[(size_t)(rt * 16 + m) * 256 + kt * 32 + k];
    });
    v16h b = load_b_frag([&](int k, int n) -> f16 {
      return (f16)Wih[(size_t)(ct * 16 + n) * 256 + kt * 32 + k];
    });
    acc = wmma16(a, b, acc);
  }
  store_c(acc, [&](int m, int n, float val) {
    xp[(size_t)(rt * 16 + m) * 192 + ct * 16 + n] =
        (f16)(val + bih[ct * 16 + n]);
  });
}

// ---------------------------------------------------------------------------
// K5: GRU recurrent scan, LDS-resident. 16 batch rows per WG, 4 waves each
// own 3 gate col-tiles; 64 sequential steps; fused tanh -> M (f16).
// ---------------------------------------------------------------------------
__global__ void __launch_bounds__(128)
gru_scan_kernel(const f16* __restrict__ xp, const float* __restrict__ Whh,
                const float* __restrict__ bhh, f16* __restrict__ M) {
  __shared__ f16 Wh[64 * 192];    // Wh[k*192+c] = Whh[c,k] (pre-transposed)
  __shared__ float Hs[16 * 64];
  __shared__ float Gh[16 * 192];
  const int tid = threadIdx.x, wave = tid >> 5;
  const int b0 = blockIdx.x * 16;
  for (int idx = tid; idx < 64 * 192; idx += 128) {
    int c = idx >> 6, k = idx & 63;
    Wh[k * 192 + c] = (f16)Whh[idx];
  }
  for (int idx = tid; idx < 16 * 64; idx += 128) Hs[idx] = 0.f;
  __syncthreads();
  for (int t = 0; t < 64; ++t) {
#pragma unroll
    for (int cc = 0; cc < 3; ++cc) {
      const int ct = wave * 3 + cc;
      v8f acc = {};
#pragma unroll
      for (int kt = 0; kt < 2; ++kt) {
        v16h a = load_a_frag([&](int m, int k) -> f16 {
          return (f16)Hs[m * 64 + kt * 32 + k];
        });
        v16h b = load_b_frag([&](int k, int n) -> f16 {
          return Wh[(kt * 32 + k) * 192 + ct * 16 + n];
        });
        acc = wmma16(a, b, acc);
      }
      store_c(acc, [&](int m, int n, float val) {
        Gh[m * 192 + ct * 16 + n] = val + bhh[ct * 16 + n];
      });
    }
    __syncthreads();
    for (int e = tid; e < 16 * 64; e += 128) {
      int i = e >> 6, j = e & 63;
      const f16* xrow = xp + ((size_t)(b0 + i) * 64 + t) * 192;
      float xr = (float)xrow[j], xz = (float)xrow[64 + j],
            xn = (float)xrow[128 + j];
      float hr = Gh[i * 192 + j], hz = Gh[i * 192 + 64 + j],
            hn = Gh[i * 192 + 128 + j];
      float r = 1.f / (1.f + __expf(-(xr + hr)));
      float z = 1.f / (1.f + __expf(-(xz + hz)));
      float nn = tanhf(xn + r * hn);
      float h = (1.f - z) * nn + z * Hs[e];
      Hs[e] = h;
      M[(size_t)(b0 + i) * 4096 + t * 64 + j] = (f16)tanhf(h);
    }
    __syncthreads();
  }
}

// ---------------------------------------------------------------------------
// K6: tiled C = A @ Bt^T GEMM for K=4096 operands (M1@M2^T, x@W^T).
// 4 waves / WG, 64x16 output tile, K staged 128-deep through LDS using
// CDNA5 async global->LDS B128 copies (ASYNCcnt), then WMMA from LDS.
// A: [*,4096] f16 row-major; Bt: [N,4096] f16 row-major (used transposed).
// ---------------------------------------------------------------------------
__global__ void __launch_bounds__(128)
gemm_nt_tiled_kernel(const f16* __restrict__ A, const f16* __restrict__ Bt,
                     float* __restrict__ C, f16* __restrict__ Ch, int ldc) {
  __shared__ f16 As[64 * 128];  // 16 KB
  __shared__ f16 Bs[16 * 128];  // 4 KB
  const int tid = threadIdx.x, w = tid >> 5;
  const int rt = blockIdx.x, ct = blockIdx.y;
  const unsigned asbase = lds_off_u32(As);
  const unsigned bsbase = lds_off_u32(Bs);
  v8f acc = {};
  for (int stage = 0; stage < 32; ++stage) {
    const int k0 = stage * 128;
    // stage A tile 64x128 (1024 x 16B transfers, 8 per thread)
#pragma unroll
    for (int pass = 0; pass < 8; ++pass) {
      int idx = pass * 128 + tid;
      int row = idx >> 4, c8 = idx & 15;
      async_copy_g2l_b128(asbase + (unsigned)(row * 128 + c8 * 8) * 2,
                          A + (size_t)(rt * 64 + row) * 4096 + k0 + c8 * 8);
    }
    // stage B tile 16x128 (256 x 16B transfers, 2 per thread)
#pragma unroll
    for (int pass = 0; pass < 2; ++pass) {
      int idx = pass * 128 + tid;
      int n = idx >> 4, c8 = idx & 15;
      async_copy_g2l_b128(bsbase + (unsigned)(n * 128 + c8 * 8) * 2,
                          Bt + (size_t)(ct * 16 + n) * 4096 + k0 + c8 * 8);
    }
    wait_async_zero();
    __syncthreads();
#pragma unroll
    for (int ks = 0; ks < 4; ++ks) {
      v16h a = load_a_frag([&](int m, int k) -> f16 {
        return As[(w * 16 + m) * 128 + ks * 32 + k];
      });
      v16h b = load_b_frag([&](int k, int n) -> f16 {
        return Bs[n * 128 + ks * 32 + k];
      });
      acc = wmma16(a, b, acc);
    }
    __syncthreads();
  }
  store_c(acc, [&](int m, int n, float val) {
    size_t idx = (size_t)(rt * 64 + w * 16 + m) * ldc + ct * 16 + n;
    C[idx] = val;
    if (Ch) Ch[idx] = (f16)val;
  });
}

// adjE = relu(P - P^T) + I; accumulate sum / sumsq
__global__ void adj_stats_kernel(const float* __restrict__ P,
                                 float* __restrict__ adjE,
                                 float* __restrict__ stats) {
  int i = blockIdx.x;
  int tid = threadIdx.x;
  float s0 = 0.f, s1 = 0.f;
  for (int j = tid; j < 1024; j += 256) {
    float v = P[(size_t)i * 1024 + j] - P[(size_t)j * 1024 + i];
    v = v > 0.f ? v : 0.f;
    if (i == j) v += 1.f;
    adjE[(size_t)i * 1024 + j] = v;
    s0 += v;
    s1 += v * v;
  }
  __shared__ float sh0[256], sh1[256];
  sh0[tid] = s0; sh1[tid] = s1;
  __syncthreads();
  for (int o = 128; o > 0; o >>= 1) {
    if (tid < o) { sh0[tid] += sh0[tid + o]; sh1[tid] += sh1[tid + o]; }
    __syncthreads();
  }
  if (tid == 0) { atomicAdd(&stats[0], sh0[0]); atomicAdd(&stats[1], sh1[0]); }
}

__global__ void thr_kernel(float* stats) {
  const float N2 = 1024.f * 1024.f;
  float mean = stats[0] / N2;
  float var = stats[1] / N2 - mean * mean;
  stats[2] = mean + 2.f * sqrtf(fmaxf(var, 0.f));
}

// mask[dst=i, src=j] = adjE[j,i] > thr   (idx_mask.T semantics)
__global__ void mask_kernel(const float* __restrict__ adjE,
                            const float* __restrict__ stats,
                            unsigned char* __restrict__ mask) {
  int i = blockIdx.x;
  float thr = stats[2];
  for (int j = threadIdx.x; j < 1024; j += 256)
    mask[(size_t)i * 1024 + j] =
        (adjE[(size_t)j * 1024 + i] > thr) ? (unsigned char)1 : (unsigned char)0;
}

__global__ void avg_half_kernel(const f16* __restrict__ A,
                                const f16* __restrict__ B,
                                f16* __restrict__ X, int n) {
  int i = blockIdx.x * blockDim.x + threadIdx.x;
  if (i < n) X[i] = (f16)(0.5f * ((float)A[i] + (float)B[i]));
}

__global__ void f32_to_f16_kernel(const float* __restrict__ src,
                                  f16* __restrict__ dst, int n) {
  int i = blockIdx.x * blockDim.x + threadIdx.x;
  if (i < n) dst[i] = (f16)src[i];
}

__global__ void gat_avec_kernel(const float* __restrict__ xw,
                                const float* __restrict__ att_src,
                                const float* __restrict__ att_dst,
                                float* __restrict__ asrc,
                                float* __restrict__ adst) {
  int t = blockIdx.x * blockDim.x + threadIdx.x;
  if (t >= 4096) return;
  int n = t >> 2, h = t & 3;
  float ss = 0.f, sd = 0.f;
  for (int c = 0; c < 64; ++c) {
    float x = xw[(size_t)n * 256 + h * 64 + c];
    ss += x * att_src[h * 64 + c];
    sd += x * att_dst[h * 64 + c];
  }
  asrc[t] = ss;
  adst[t] = sd;
}

__global__ void gat_stats_kernel(const unsigned char* __restrict__ mask,
                                 const float* __restrict__ asrc,
                                 const float* __restrict__ adst,
                                 float* __restrict__ rmax,
                                 float* __restrict__ rsum) {
  int t = blockIdx.x * blockDim.x + threadIdx.x;
  if (t >= 4096) return;
  int n = t >> 2, h = t & 3;
  float ad = adst[t];
  float mx = -3.0e38f;
  for (int m = 0; m < 1024; ++m)
    if (mask[(size_t)n * 1024 + m]) {
      float e = ad + asrc[m * 4 + h];
      e = e < 0.f ? 0.2f * e : e;
      mx = fmaxf(mx, e);
    }
  float sm = 0.f;
  for (int m = 0; m < 1024; ++m)
    if (mask[(size_t)n * 1024 + m]) {
      float e = ad + asrc[m * 4 + h];
      e = e < 0.f ? 0.2f * e : e;
      sm += __expf(e - mx);
    }
  rmax[t] = mx;
  rsum[t] = sm;
}

// WMMA aggregation out[n,h,:] = sum_m alpha[n,m,h] * xw[m,h,:]
__global__ void gat_agg_kernel(const unsigned char* __restrict__ mask,
                               const float* __restrict__ asrc,
                               const float* __restrict__ adst,
                               const float* __restrict__ rmax,
                               const float* __restrict__ rsum,
                               const f16* __restrict__ xwh,
                               float* __restrict__ gout) {
  const int nt = blockIdx.x;  // 64 node tiles
  const int h = blockIdx.y;   // 4 heads
  __shared__ f16 Al[16 * 32];
  const int lane = threadIdx.x;
  v8f zero = {};
  v8f acc[4] = {zero, zero, zero, zero};
  for (int kt = 0; kt < 32; ++kt) {
    for (int e = lane; e < 512; e += 32) {
      int i = e >> 5, k = e & 31;
      int n = nt * 16 + i, m = kt * 32 + k;
      float al = 0.f;
      if (mask[(size_t)n * 1024 + m]) {
        float ev = adst[n * 4 + h] + asrc[m * 4 + h];
        ev = ev < 0.f ? 0.2f * ev : ev;
        al = __expf(ev - rmax[n * 4 + h]) / rsum[n * 4 + h];
      }
      Al[e] = (f16)al;
    }
    __syncthreads();
    v16h a = load_a_frag([&](int mm, int k) -> f16 { return Al[mm * 32 + k]; });
#pragma unroll
    for (int ct = 0; ct < 4; ++ct) {
      v16h b = load_b_frag([&](int k, int n2) -> f16 {
        return xwh[(size_t)(kt * 32 + k) * 256 + h * 64 + ct * 16 + n2];
      });
      acc[ct] = wmma16(a, b, acc[ct]);
    }
    __syncthreads();
  }
#pragma unroll
  for (int ct = 0; ct < 4; ++ct)
    store_c(acc[ct], [&](int m, int n, float val) {
      gout[(size_t)(nt * 16 + m) * 256 + h * 64 + ct * 16 + n] = val;
    });
}

// ---------------------------------------------------------------------------
// Tail: combine + pairnorm + prediction
// ---------------------------------------------------------------------------
__global__ void combine_kernel(const float* __restrict__ g1,
                               const float* __restrict__ b1,
                               const float* __restrict__ g2,
                               const float* __restrict__ b2,
                               const float* __restrict__ alpha,
                               const float* __restrict__ decay,
                               float* __restrict__ comb) {
  int idx = blockIdx.x * blockDim.x + threadIdx.x;
  if (idx >= 1024 * 256) return;
  int c = idx & 255;
  comb[idx] = (g1[idx] + b1[c]) * alpha[0] + (g2[idx] + b2[c]) * decay[0];
}

__global__ void colmean_kernel(const float* __restrict__ comb,
                               float* __restrict__ colmean) {
  int c = blockIdx.x;
  int tid = threadIdx.x;
  float s = 0.f;
  for (int r = tid; r < 1024; r += 256) s += comb[(size_t)r * 256 + c];
  __shared__ float sh[256];
  sh[tid] = s;
  __syncthreads();
  for (int o = 128; o > 0; o >>= 1) {
    if (tid < o) sh[tid] += sh[tid + o];
    __syncthreads();
  }
  if (tid == 0) colmean[c] = sh[0] * (1.f / 1024.f);
}

__global__ void pairnorm_kernel(const float* __restrict__ comb,
                                const float* __restrict__ colmean,
                                f16* __restrict__ embn) {
  int r = blockIdx.x;
  int tid = threadIdx.x;
  float y = comb[(size_t)r * 256 + tid] - colmean[tid];
  __shared__ float sh[256];
  __shared__ float inv;
  sh[tid] = y * y;
  __syncthreads();
  for (int o = 128; o > 0; o >>= 1) {
    if (tid < o) sh[tid] += sh[tid + o];
    __syncthreads();
  }
  if (tid == 0) inv = 1.f / sqrtf(1e-6f + sh[0]);
  __syncthreads();
  embn[(size_t)r * 256 + tid] = (f16)(y * inv);
}

__global__ void pred_kernel(const f16* __restrict__ E,
                            const float* __restrict__ Wp,
                            const float* __restrict__ bp,
                            float* __restrict__ out) {
  const int rt = blockIdx.x, ct = blockIdx.y;  // 64 x 16
  v8f acc = {};
  for (int kt = 0; kt < 8; ++kt) {
    v16h a = load_a_frag([&](int m, int k) -> f16 {
      return E[(size_t)(rt * 16 + m) * 256 + kt * 32 + k];
    });
    v16h b = load_b_frag([&](int k, int n) -> f16 {
      return (f16)Wp[(size_t)(ct * 16 + n) * 256 + kt * 32 + k];
    });
    acc = wmma16(a, b, acc);
  }
  store_c(acc, [&](int m, int n, float val) {
    val += bp[ct * 16 + n];
    out[(size_t)(rt * 16 + m) * 256 + ct * 16 + n] = 1.f / (1.f + __expf(-val));
  });
}

__global__ void zero_kernel(float* p, int n) {
  int i = blockIdx.x * blockDim.x + threadIdx.x;
  if (i < n) p[i] = 0.f;
}

// ---------------------------------------------------------------------------
// Orchestration
// ---------------------------------------------------------------------------
extern "C" void kernel_launch(void* const* d_in, const int* in_sizes, int n_in,
                              void* d_out, int out_size, void* d_ws,
                              size_t ws_size, hipStream_t stream) {
  (void)in_sizes; (void)n_in; (void)out_size; (void)ws_size;
  const float* data  = (const float*)d_in[0];
  const float* hs1   = (const float*)d_in[1];
  const float* Wqkv  = (const float*)d_in[2];
  const float* bqkv  = (const float*)d_in[3];
  const float* Wo    = (const float*)d_in[4];
  const float* bo    = (const float*)d_in[5];
  const float* gWih[3] = {(const float*)d_in[6], (const float*)d_in[10],
                          (const float*)d_in[14]};
  const float* gWhh[3] = {(const float*)d_in[7], (const float*)d_in[11],
                          (const float*)d_in[15]};
  const float* gbih[3] = {(const float*)d_in[8], (const float*)d_in[12],
                          (const float*)d_in[16]};
  const float* gbhh[3] = {(const float*)d_in[9], (const float*)d_in[13],
                          (const float*)d_in[17]};
  const float* gatW[2]  = {(const float*)d_in[18], (const float*)d_in[22]};
  const float* gatAS[2] = {(const float*)d_in[19], (const float*)d_in[23]};
  const float* gatAD[2] = {(const float*)d_in[20], (const float*)d_in[24]};
  const float* gatB[2]  = {(const float*)d_in[21], (const float*)d_in[25]};
  const float* predW = (const float*)d_in[26];
  const float* predB = (const float*)d_in[27];
  const float* alpha = (const float*)d_in[28];
  const float* decay = (const float*)d_in[29];
  float* out = (float*)d_out;

  // ---- workspace arena (lifetime-aliased) ----
  char* ws = (char*)d_ws;
  size_t off = 0;
  auto arena = [&](size_t bytes) -> char* {
    char* p = ws + off;
    off += (bytes + 255) & ~(size_t)255;
    return p;
  };
  const size_t QKV_B = (size_t)16777216 * 2;  // 1024 bmh * 1024 s * 16 d, f16
  f16* Qh   = (f16*)arena(QKV_B);
  f16* Kh   = (f16*)arena(QKV_B);
  f16* Vh   = (f16*)arena(QKV_B);
  f16* Oh   = (f16*)arena(QKV_B);   // [S*Bm, 64] f16
  f16* emb1 = (f16*)arena(QKV_B);   // [1024,64,256] f16
  f16* emb2 = (f16*)arena(QKV_B);
  // aliases: xprojs reuse Q/K/V (dead after attention); M reuse Oh
  f16* xp[3] = {Qh, Kh, Vh};        // each 25.2MB <= 33.6MB
  f16* M1 = Oh;                     // 3 * 8.4MB <= 33.6MB
  f16* M2 = M1 + 4194304;
  f16* M3 = M2 + 4194304;
  float*         P     = (float*)arena((size_t)1048576 * 4);
  float*         adjE  = (float*)arena((size_t)1048576 * 4);
  unsigned char* mask1 = (unsigned char*)arena(1048576);
  unsigned char* mask2 = (unsigned char*)arena(1048576);
  f16*   x1h  = (f16*)arena((size_t)4194304 * 2);
  f16*   x2h  = (f16*)arena((size_t)4194304 * 2);
  f16*   Wg16 = (f16*)arena((size_t)1048576 * 2);   // f16 copy of GAT weight
  float* xwf  = (float*)arena((size_t)262144 * 4);
  f16*   xwh  = (f16*)arena((size_t)262144 * 2);
  float* asrc = (float*)arena(4096 * 4);
  float* adst = (float*)arena(4096 * 4);
  float* rmax = (float*)arena(4096 * 4);
  float* rsum = (float*)arena(4096 * 4);
  float* gout1 = (float*)arena((size_t)262144 * 4);
  float* gout2 = (float*)arena((size_t)262144 * 4);
  float* comb  = (float*)arena((size_t)262144 * 4);
  float* cmean = (float*)arena(256 * 4);
  f16*   embn  = (f16*)arena((size_t)262144 * 2);
  float* stats = (float*)arena(64);

  // ---- MHA on data -> emb1, on hs1 -> emb2 ----
  qkv_proj_kernel<<<dim3(16384, 12), 32, 0, stream>>>(data, Wqkv, bqkv, Qh, Kh, Vh);
  flash_attn_kernel<<<dim3(1024, 64), 32, 0, stream>>>(Qh, Kh, Vh, Oh);
  attn_outproj_kernel<<<dim3(16384, 4), 32, 0, stream>>>(Oh, Wo, bo, emb1);
  qkv_proj_kernel<<<dim3(16384, 12), 32, 0, stream>>>(hs1, Wqkv, bqkv, Qh, Kh, Vh);
  flash_attn_kernel<<<dim3(1024, 64), 32, 0, stream>>>(Qh, Kh, Vh, Oh);
  attn_outproj_kernel<<<dim3(16384, 4), 32, 0, stream>>>(Oh, Wo, bo, emb2);

  // ---- GRUs: g1,g2 from emb1; g3 from emb2; fused tanh -> M ----
  gru_xproj_kernel<<<dim3(4096, 12), 32, 0, stream>>>(emb1, gWih[0], gbih[0], xp[0]);
  gru_xproj_kernel<<<dim3(4096, 12), 32, 0, stream>>>(emb1, gWih[1], gbih[1], xp[1]);
  gru_xproj_kernel<<<dim3(4096, 12), 32, 0, stream>>>(emb2, gWih[2], gbih[2], xp[2]);
  gru_scan_kernel<<<64, 128, 0, stream>>>(xp[0], gWhh[0], gbhh[0], M1);
  gru_scan_kernel<<<64, 128, 0, stream>>>(xp[1], gWhh[1], gbhh[1], M2);
  gru_scan_kernel<<<64, 128, 0, stream>>>(xp[2], gWhh[2], gbhh[2], M3);

  const f16* Mb[2] = {M2, M3};
  f16* xh[2] = {x1h, x2h};
  unsigned char* masks[2] = {mask1, mask2};
  float* gouts[2] = {gout1, gout2};
  for (int g = 0; g < 2; ++g) {
    // adjacency: P = M1 @ Mb^T (async-LDS tiled WMMA GEMM);
    // adjE = relu(P - P^T) + I; threshold; mask
    gemm_nt_tiled_kernel<<<dim3(16, 64), 128, 0, stream>>>(M1, Mb[g], P,
                                                           (f16*)nullptr, 1024);
    zero_kernel<<<1, 64, 0, stream>>>(stats, 4);
    adj_stats_kernel<<<1024, 256, 0, stream>>>(P, adjE, stats);
    thr_kernel<<<1, 1, 0, stream>>>(stats);
    mask_kernel<<<1024, 256, 0, stream>>>(adjE, stats, masks[g]);
    // GAT
    avg_half_kernel<<<16384, 256, 0, stream>>>(M1, Mb[g], xh[g], 4194304);
    f32_to_f16_kernel<<<4096, 256, 0, stream>>>(gatW[g], Wg16, 1048576);
    gemm_nt_tiled_kernel<<<dim3(16, 16), 128, 0, stream>>>(xh[g], Wg16, xwf,
                                                           xwh, 256);
    gat_avec_kernel<<<16, 256, 0, stream>>>(xwf, gatAS[g], gatAD[g], asrc, adst);
    gat_stats_kernel<<<16, 256, 0, stream>>>(masks[g], asrc, adst, rmax, rsum);
    gat_agg_kernel<<<dim3(64, 4), 32, 0, stream>>>(masks[g], asrc, adst, rmax,
                                                   rsum, xwh, gouts[g]);
  }

  // ---- combine, pairnorm, prediction ----
  combine_kernel<<<1024, 256, 0, stream>>>(gout1, gatB[0], gout2, gatB[1],
                                           alpha, decay, comb);
  colmean_kernel<<<256, 256, 0, stream>>>(comb, cmean);
  pairnorm_kernel<<<1024, 256, 0, stream>>>(comb, cmean, embn);
  pred_kernel<<<dim3(64, 16), 32, 0, stream>>>(embn, predW, predB, out);
}
